// MyMutualMultiAttention_34351148434041
// MI455X (gfx1250) — matrix operand order repaired and checked by reading, not compiled, not run on hardware
//
#include <hip/hip_runtime.h>

typedef __attribute__((ext_vector_type(16))) _Float16 v16h;
typedef __attribute__((ext_vector_type(8)))  _Float16 v8h;
typedef __attribute__((ext_vector_type(4)))  _Float16 v4h;
typedef __attribute__((ext_vector_type(8)))  float    v8f;
typedef __attribute__((ext_vector_type(4)))  float    v4f;
typedef __attribute__((ext_vector_type(4)))  int      v4i;

#define B_   2
#define S_   2048
#define E_   1024
#define H_   16
#define D_   64
#define HD_  1024
#define SCALE_ 0.125f          // 1/sqrt(64)
#define MYINF_ 1e13f

// ---- optional CDNA5 async global->LDS path (ASYNCcnt) ----
#if __has_builtin(__builtin_amdgcn_global_load_async_to_lds_b128)
#define HAVE_ASYNC_LDS 1
#define GLOBAL_AS __attribute__((address_space(1)))
#define LDS_AS    __attribute__((address_space(3)))
__device__ __forceinline__ void async_cp16(const void* gsrc, void* ldst) {
  __builtin_amdgcn_global_load_async_to_lds_b128(
      (GLOBAL_AS v4i*)gsrc, (LDS_AS v4i*)ldst, 0, 0);
}
__device__ __forceinline__ void wait_async0() {
#if __has_builtin(__builtin_amdgcn_s_wait_asynccnt)
  __builtin_amdgcn_s_wait_asynccnt(0);
#else
  asm volatile("s_wait_asynccnt 0x0" ::: "memory");
#endif
}
#else
#define HAVE_ASYNC_LDS 0
__device__ __forceinline__ void wait_async0() {}
#endif

// ---- WMMA fragment loaders (contiguous k order: k = 16*(lane/16) + e, wave32) ----
// All fragments are built from two contiguous 16B LDS reads.

__device__ __forceinline__ v16h frag_from2(const _Float16* p0, const _Float16* p1) {
  v8h lo = *(const v8h*)p0;
  v8h hi = *(const v8h*)p1;
  return __builtin_shufflevector(lo, hi, 0, 1, 2, 3, 4, 5, 6, 7,
                                 8, 9, 10, 11, 12, 13, 14, 15);
}

// A 16x32 f16 from LDS stored [row][k] (stride in halves, multiple of 8):
// lane L: row=L&15, elems 0..15 = K kb..kb+15 (kb = 16*(L>=16))
__device__ __forceinline__ v16h ldsA16x32(const _Float16* lds, int row0, int k0, int stride) {
  const int lane = threadIdx.x & 31;
  const _Float16* base = lds + (row0 + (lane & 15)) * stride + k0 + (lane >> 4) * 16;
  return frag_from2(base, base + 8);
}

// B 32x16 f16 from LDS stored [n][k] (K-contiguous): lane L holds col N=L&15,
// K = 16*(L>=16) + e, e=0..15  -> one 32B run.
__device__ __forceinline__ v16h ldsB32x16_nk(const _Float16* lds, int k0, int col0, int stride) {
  const int lane = threadIdx.x & 31;
  const _Float16* base = lds + (col0 + (lane & 15)) * stride + k0 + (lane >> 4) * 16;
  return frag_from2(base, base + 8);
}

// ---- GEMM tile staging (counted loops: no trip-count branches) ----

// A tile 128x32 fp32 -> f16 LDS [row][k] stride 40
__device__ __forceinline__ void stage_A_f32(const float* __restrict__ A,
                                            _Float16* dst, int m0, int k0, int t) {
#pragma unroll
  for (int j = 0; j < 4; ++j) {
    const int i = t + j * 256;
    const int r = i >> 3, k4 = (i & 7) * 4;
    const v4f s = *(const v4f*)&A[(size_t)(m0 + r) * E_ + k0 + k4];
    v4h d; d[0] = (_Float16)s[0]; d[1] = (_Float16)s[1];
           d[2] = (_Float16)s[2]; d[3] = (_Float16)s[3];
    *(v4h*)&dst[r * 40 + k4] = d;
  }
}

// W tile 32x128 fp32 -> f16 LDS transposed [n][k] stride 40
__device__ __forceinline__ void stage_B_f32(const float* __restrict__ W,
                                            _Float16* dst, int n0, int k0, int t) {
#pragma unroll
  for (int j = 0; j < 4; ++j) {
    const int i = t + j * 256;
    const int k = i >> 5, n4 = (i & 31) * 4;
    const v4f s = *(const v4f*)&W[(size_t)(k0 + k) * HD_ + n0 + n4];
    dst[(n4 + 0) * 40 + k] = (_Float16)s[0];
    dst[(n4 + 1) * 40 + k] = (_Float16)s[1];
    dst[(n4 + 2) * 40 + k] = (_Float16)s[2];
    dst[(n4 + 3) * 40 + k] = (_Float16)s[3];
  }
}

// ---------------- Kernel 1: fused QKV projection ----------------
// z=0: Q = dec@Wq+bq ; z=1: K = enc@Wk+bk ; z=2: V = enc@Wv+bv
// C[4096,1024] tiles of 128x128, BK=32, double-buffered LDS, 8 waves x 32x64.
__global__ __launch_bounds__(256) void qkv_proj_kernel(
    const float* __restrict__ enc, const float* __restrict__ dec,
    const float* __restrict__ Wq, const float* __restrict__ Wk, const float* __restrict__ Wv,
    const float* __restrict__ bq, const float* __restrict__ bk, const float* __restrict__ bv,
    _Float16* __restrict__ Qb, _Float16* __restrict__ Kb, _Float16* __restrict__ Vb) {
  __shared__ _Float16 Ash[2][128 * 40];
  __shared__ _Float16 Bsh[2][128 * 40];

  const int t = threadIdx.x;
  const int z = blockIdx.z;
  const float* A    = (z == 0) ? dec : enc;
  const float* W    = (z == 0) ? Wq : (z == 1) ? Wk : Wv;
  const float* bias = (z == 0) ? bq : (z == 1) ? bk : bv;
  _Float16*    Out  = (z == 0) ? Qb : (z == 1) ? Kb : Vb;

  const int m0 = blockIdx.y * 128;
  const int n0 = blockIdx.x * 128;
  const int wave = t >> 5, lane = t & 31;
  const int wm = wave >> 1, wn = wave & 1;          // 4x2 wave grid
  const int lc = lane & 15, half = lane >> 4;

  v8f acc[2][4];
#pragma unroll
  for (int mi = 0; mi < 2; ++mi)
#pragma unroll
    for (int ni = 0; ni < 4; ++ni) acc[mi][ni] = {};

  stage_A_f32(A, Ash[0], m0, 0, t);
  stage_B_f32(W, Bsh[0], n0, 0, t);
  __syncthreads();

  for (int kt = 0; kt < 32; ++kt) {
    const int cur = kt & 1;
    if (kt < 31) {                        // prefetch next tile into other buffer
      stage_A_f32(A, Ash[cur ^ 1], m0, (kt + 1) * 32, t);
      stage_B_f32(W, Bsh[cur ^ 1], n0, (kt + 1) * 32, t);
    }
    v16h bf[4];
#pragma unroll
    for (int ni = 0; ni < 4; ++ni)
      bf[ni] = ldsB32x16_nk(Bsh[cur], 0, wn * 64 + ni * 16, 40);
#pragma unroll
    for (int mi = 0; mi < 2; ++mi) {
      v16h a = ldsA16x32(Ash[cur], wm * 32 + mi * 16, 0, 40);
#pragma unroll
      for (int ni = 0; ni < 4; ++ni)
        acc[mi][ni] = __builtin_amdgcn_wmma_f32_16x16x32_f16(
            false, a, false, bf[ni], (short)0, acc[mi][ni], false, false);
    }
    __syncthreads();
  }

#pragma unroll
  for (int ni = 0; ni < 4; ++ni) {
    const int col = n0 + wn * 64 + ni * 16 + lc;
    const float bcol = bias[col];
#pragma unroll
    for (int mi = 0; mi < 2; ++mi)
#pragma unroll
      for (int r = 0; r < 8; ++r) {
        const int row = m0 + wm * 32 + mi * 16 + r + 8 * half;
        Out[(size_t)row * HD_ + col] = (_Float16)(acc[mi][ni][r] + bcol);
      }
  }
}

// ---------------- Kernel 2: fused flash attention ----------------
// One workgroup = 128 query rows of one (b,h); 8 waves x 16 rows.
// Double-buffered K/V tiles: async prefetch of tile kt+1 overlaps WMMA on kt.
__global__ __launch_bounds__(256) void attention_kernel(
    const _Float16* __restrict__ Qb, const _Float16* __restrict__ Kb,
    const _Float16* __restrict__ Vb,
    const int* __restrict__ enc_pad, const int* __restrict__ dec_pad,
    _Float16* __restrict__ Ctx) {
  __shared__ _Float16 Qs[128 * 72];      // [qrow][d]
  __shared__ _Float16 Ks[2][64 * 72];    // [key][d]   (B for Q*K^T)
  __shared__ _Float16 Vt[2][64 * 72];    // [d][key]   (B for P*V)
  __shared__ _Float16 Ps[128 * 72];      // [qrow][key] per-wave P scratch
  __shared__ int mq[128];
  __shared__ int mk[2][64];

  const int t = threadIdx.x, wave = t >> 5, lane = t & 31;
  const int lc = lane & 15, half = lane >> 4;
  const int q0 = blockIdx.x * 128;
  const int h  = blockIdx.y;
  const int b  = blockIdx.z;

  auto stage_kv = [&](int kt, int buf) {
    const int kb = kt * 64;
    const size_t baseK = ((size_t)b * S_ + kb) * HD_ + (size_t)h * D_;
#if HAVE_ASYNC_LDS
#pragma unroll
    for (int j = 0; j < 2; ++j) {
      const int i = t + j * 256;
      const int r = i >> 3, d8 = (i & 7) * 8;
      async_cp16(&Kb[baseK + (size_t)r * HD_ + d8], &Ks[buf][r * 72 + d8]);
    }
#else
#pragma unroll
    for (int j = 0; j < 2; ++j) {
      const int i = t + j * 256;
      const int r = i >> 3, d8 = (i & 7) * 8;
      *(v8h*)&Ks[buf][r * 72 + d8] = *(const v8h*)&Kb[baseK + (size_t)r * HD_ + d8];
    }
#endif
    // V tile transposed while staging -> Vt[d][key]
#pragma unroll
    for (int j = 0; j < 2; ++j) {
      const int i = t + j * 256;
      const int r = i >> 3, d8 = (i & 7) * 8;
      const v8h v = *(const v8h*)&Vb[baseK + (size_t)r * HD_ + d8];
#pragma unroll
      for (int jj = 0; jj < 8; ++jj) Vt[buf][(d8 + jj) * 72 + r] = v[jj];
    }
    if (t < 64) mk[buf][t] = enc_pad[b * S_ + kb + t];
  };

  // ---- prologue: stage Q + tile 0 ----
  const size_t baseQ = ((size_t)b * S_ + q0) * HD_ + (size_t)h * D_;
#if HAVE_ASYNC_LDS
#pragma unroll
  for (int j = 0; j < 4; ++j) {
    const int i = t + j * 256;
    const int r = i >> 3, d8 = (i & 7) * 8;
    async_cp16(&Qb[baseQ + (size_t)r * HD_ + d8], &Qs[r * 72 + d8]);
  }
#else
#pragma unroll
  for (int j = 0; j < 4; ++j) {
    const int i = t + j * 256;
    const int r = i >> 3, d8 = (i & 7) * 8;
    *(v8h*)&Qs[r * 72 + d8] = *(const v8h*)&Qb[baseQ + (size_t)r * HD_ + d8];
  }
#endif
  stage_kv(0, 0);
  if (t < 128) mq[t] = dec_pad[b * S_ + q0 + t];
  wait_async0();
  __syncthreads();
  int rowm[8];
#pragma unroll
  for (int r = 0; r < 8; ++r) rowm[r] = mq[wave * 16 + r + 8 * half];

  v8f o[4];
#pragma unroll
  for (int ni = 0; ni < 4; ++ni) o[ni] = {};
  float m[8], l[8];
#pragma unroll
  for (int r = 0; r < 8; ++r) { m[r] = -__builtin_inff(); l[r] = 0.0f; }

  for (int kt = 0; kt < S_ / 64; ++kt) {
    const int cur = kt & 1;
    if (kt + 1 < S_ / 64) stage_kv(kt + 1, cur ^ 1);   // overlap with compute

    // ---- S = Q * K^T  (16x64 per wave, k-dim 64 = 2 WMMAs) ----
    v8f sa[4];
#pragma unroll
    for (int ni = 0; ni < 4; ++ni) sa[ni] = {};
#pragma unroll
    for (int kk = 0; kk < 2; ++kk) {
      v16h a = ldsA16x32(Qs, wave * 16, kk * 32, 72);
#pragma unroll
      for (int ni = 0; ni < 4; ++ni) {
        v16h bf = ldsB32x16_nk(Ks[cur], kk * 32, ni * 16, 72);
        sa[ni] = __builtin_amdgcn_wmma_f32_16x16x32_f16(
            false, a, false, bf, (short)0, sa[ni], false, false);
      }
    }

    // ---- scale + replace-mask ----
    int colm[4];
#pragma unroll
    for (int ni = 0; ni < 4; ++ni) colm[ni] = mk[cur][ni * 16 + lc];
    float sv[4][8];
#pragma unroll
    for (int ni = 0; ni < 4; ++ni)
#pragma unroll
      for (int r = 0; r < 8; ++r)
        sv[ni][r] = (colm[ni] | rowm[r]) ? -MYINF_ : sa[ni][r] * SCALE_;

    // ---- online softmax: row max across 64 cols (4 tiles + 16-lane half) ----
    float mnew[8], corr[8], rs[8];
#pragma unroll
    for (int r = 0; r < 8; ++r) {
      float mx = fmaxf(fmaxf(sv[0][r], sv[1][r]), fmaxf(sv[2][r], sv[3][r]));
      mx = fmaxf(mx, __shfl_xor(mx, 8, 32));
      mx = fmaxf(mx, __shfl_xor(mx, 4, 32));
      mx = fmaxf(mx, __shfl_xor(mx, 2, 32));
      mx = fmaxf(mx, __shfl_xor(mx, 1, 32));
      mnew[r] = fmaxf(m[r], mx);
      corr[r] = __expf(m[r] - mnew[r]);
      m[r] = mnew[r];
      rs[r] = 0.0f;
    }

    // ---- P = exp(S - m), spill to per-wave LDS rows (same-wave, in-order) ----
#pragma unroll
    for (int ni = 0; ni < 4; ++ni)
#pragma unroll
      for (int r = 0; r < 8; ++r) {
        const float p = __expf(sv[ni][r] - mnew[r]);
        rs[r] += p;
        Ps[(wave * 16 + r + 8 * half) * 72 + ni * 16 + lc] = (_Float16)p;
      }
#pragma unroll
    for (int r = 0; r < 8; ++r) {
      float s = rs[r];
      s += __shfl_xor(s, 8, 32);
      s += __shfl_xor(s, 4, 32);
      s += __shfl_xor(s, 2, 32);
      s += __shfl_xor(s, 1, 32);
      l[r] = l[r] * corr[r] + s;
    }

    // ---- rescale running O, then O += P * V ----
#pragma unroll
    for (int ni = 0; ni < 4; ++ni)
#pragma unroll
      for (int r = 0; r < 8; ++r) o[ni][r] *= corr[r];

#pragma unroll
    for (int kk = 0; kk < 2; ++kk) {
      v16h a = ldsA16x32(Ps, wave * 16, kk * 32, 72);
#pragma unroll
      for (int ni = 0; ni < 4; ++ni) {
        v16h bf = ldsB32x16_nk(Vt[cur], kk * 32, ni * 16, 72);
        o[ni] = __builtin_amdgcn_wmma_f32_16x16x32_f16(
            false, a, false, bf, (short)0, o[ni], false, false);
      }
    }

    wait_async0();       // next tile landed (this wave's issues)
    __syncthreads();     // all waves done with cur; next buffer ready
  }

  // ---- normalize + store ctx (f16, [B,S,H*D]) ----
  const size_t baseO = ((size_t)b * S_ + q0) * HD_ + (size_t)h * D_;
#pragma unroll
  for (int r = 0; r < 8; ++r) {
    const float inv = 1.0f / l[r];
    const int row = wave * 16 + r + 8 * half;
#pragma unroll
    for (int ni = 0; ni < 4; ++ni)
      Ctx[baseO + (size_t)row * HD_ + ni * 16 + lc] = (_Float16)(o[ni][r] * inv);
  }
}

// ---------------- Kernel 3: output projection ----------------
// out[4096,1024] = ctx(f16) @ Wo + bo, fp32 result. Double-buffered; A tile async.
__global__ __launch_bounds__(256) void out_proj_kernel(
    const _Float16* __restrict__ Ctx, const float* __restrict__ Wo,
    const float* __restrict__ bo, float* __restrict__ Out) {
  __shared__ _Float16 Ash[2][128 * 40];
  __shared__ _Float16 Bsh[2][128 * 40];

  const int t = threadIdx.x;
  const int m0 = blockIdx.y * 128;
  const int n0 = blockIdx.x * 128;
  const int wave = t >> 5, lane = t & 31;
  const int wm = wave >> 1, wn = wave & 1;
  const int lc = lane & 15, half = lane >> 4;

  auto stage_A = [&](int k0, int buf) {
#if HAVE_ASYNC_LDS
#pragma unroll
    for (int j = 0; j < 2; ++j) {
      const int i = t + j * 256;
      const int r = i >> 2, k8 = (i & 3) * 8;
      async_cp16(&Ctx[(size_t)(m0 + r) * HD_ + k0 + k8], &Ash[buf][r * 40 + k8]);
    }
#else
#pragma unroll
    for (int j = 0; j < 2; ++j) {
      const int i = t + j * 256;
      const int r = i >> 2, k8 = (i & 3) * 8;
      *(v8h*)&Ash[buf][r * 40 + k8] = *(const v8h*)&Ctx[(size_t)(m0 + r) * HD_ + k0 + k8];
    }
#endif
  };

  v8f acc[2][4];
#pragma unroll
  for (int mi = 0; mi < 2; ++mi)
#pragma unroll
    for (int ni = 0; ni < 4; ++ni) acc[mi][ni] = {};

  stage_A(0, 0);
  stage_B_f32(Wo, Bsh[0], n0, 0, t);
  wait_async0();
  __syncthreads();

  for (int kt = 0; kt < 32; ++kt) {
    const int cur = kt & 1;
    if (kt < 31) {
      stage_A((kt + 1) * 32, cur ^ 1);
      stage_B_f32(Wo, Bsh[cur ^ 1], n0, (kt + 1) * 32, t);
    }
    v16h bf[4];
#pragma unroll
    for (int ni = 0; ni < 4; ++ni)
      bf[ni] = ldsB32x16_nk(Bsh[cur], 0, wn * 64 + ni * 16, 40);
#pragma unroll
    for (int mi = 0; mi < 2; ++mi) {
      v16h a = ldsA16x32(Ash[cur], wm * 32 + mi * 16, 0, 40);
#pragma unroll
      for (int ni = 0; ni < 4; ++ni)
        acc[mi][ni] = __builtin_amdgcn_wmma_f32_16x16x32_f16(
            false, a, false, bf[ni], (short)0, acc[mi][ni], false, false);
    }
    wait_async0();
    __syncthreads();
  }

#pragma unroll
  for (int ni = 0; ni < 4; ++ni) {
    const int col = n0 + wn * 64 + ni * 16 + lc;
    const float bcol = bo[col];
#pragma unroll
    for (int mi = 0; mi < 2; ++mi)
#pragma unroll
      for (int r = 0; r < 8; ++r) {
        const int row = m0 + wm * 32 + mi * 16 + r + 8 * half;
        Out[(size_t)row * E_ + col] = acc[mi][ni][r] + bcol;
      }
  }
}

// ---------------- host launcher ----------------
extern "C" void kernel_launch(void* const* d_in, const int* in_sizes, int n_in,
                              void* d_out, int out_size, void* d_ws, size_t ws_size,
                              hipStream_t stream) {
  const float* enc     = (const float*)d_in[0];
  const float* dec     = (const float*)d_in[1];
  const int*   enc_pad = (const int*)d_in[2];
  const int*   dec_pad = (const int*)d_in[3];
  const float* Wq = (const float*)d_in[4];
  const float* bq = (const float*)d_in[5];
  const float* Wk = (const float*)d_in[6];
  const float* bk = (const float*)d_in[7];
  const float* Wv = (const float*)d_in[8];
  const float* bv = (const float*)d_in[9];
  const float* Wo = (const float*)d_in[10];
  const float* bo = (const float*)d_in[11];
  float* out = (float*)d_out;

  // workspace: Q, K, V, Ctx each B*S*HD f16 (8 MB) -> 32 MB total
  _Float16* ws  = (_Float16*)d_ws;
  const size_t tok = (size_t)B_ * S_ * HD_;
  _Float16* Qb  = ws;
  _Float16* Kb  = Qb + tok;
  _Float16* Vb  = Kb + tok;
  _Float16* Ctx = Vb + tok;

  qkv_proj_kernel<<<dim3(HD_ / 128, (B_ * S_) / 128, 3), 256, 0, stream>>>(
      enc, dec, Wq, Wk, Wv, bq, bk, bv, Qb, Kb, Vb);
  attention_kernel<<<dim3(S_ / 128, H_, B_), 256, 0, stream>>>(
      Qb, Kb, Vb, enc_pad, dec_pad, Ctx);
  out_proj_kernel<<<dim3(E_ / 128, (B_ * S_) / 128), 256, 0, stream>>>(
      Ctx, Wo, bo, out);
}
